// MultiHeadAttention_34875134443653
// MI455X (gfx1250) — compile-verified, matrix operand order checked by reading, not run on hardware
//
#include <hip/hip_runtime.h>
#include <hip/hip_bf16.h>

// ---------------------------------------------------------------------------
// MHA block for MI455X (gfx1250): bf16 WMMA, TDM-staged GEMMs, async-LDS attn.
// ---------------------------------------------------------------------------

#define Bsz 4
#define Hn  16
#define Ssz 2048
#define Dh  64
#define Dm  1024

typedef __bf16 bf16;
typedef bf16  v16bf __attribute__((ext_vector_type(16)));
typedef bf16  v8bf  __attribute__((ext_vector_type(8)));
typedef float v8f   __attribute__((ext_vector_type(8)));
typedef float v4f   __attribute__((ext_vector_type(4)));

typedef unsigned int uint4v __attribute__((ext_vector_type(4)));
typedef int          int8v  __attribute__((ext_vector_type(8)));
typedef int          int4v  __attribute__((ext_vector_type(4)));

// D = A(16x32 bf16) * B(32x16 bf16) + C(16x16 f32)
__device__ __forceinline__ v8f wmma_bf16(v16bf a, v16bf b, v8f c) {
  return __builtin_amdgcn_wmma_f32_16x16x32_bf16(
      false, a, false, b, (short)0, c, false, false);
}

// Async copy: 16B per lane, global -> LDS, tracked by ASYNCcnt.
__device__ __forceinline__ void async_b128(unsigned lds_addr, const void* gaddr) {
  asm volatile("global_load_async_to_lds_b128 %0, %1, off"
               :: "v"(lds_addr), "v"(gaddr) : "memory");
}
#define WAIT_ASYNC(N) asm volatile("s_wait_asynccnt %0" :: "n"(N) : "memory")

#if __has_builtin(__builtin_amdgcn_s_wait_tensorcnt)
#define WAIT_TENSOR(N) __builtin_amdgcn_s_wait_tensorcnt((short)(N))
#else
#define WAIT_TENSOR(N) asm volatile("s_wait_tensorcnt %0" :: "n"(N) : "memory")
#endif

#if __has_builtin(__builtin_amdgcn_tensor_load_to_lds)
#define HAS_TDM 1
#else
#define HAS_TDM 0
#endif

__device__ __forceinline__ unsigned lds_off(const void* p) {
  return (unsigned)(size_t)p;   // generic->LDS aperture: low 32 bits = offset
}

#if HAS_TDM
// One TDM 2D tile load: tile_w x tile_h elements (2B each) from a row-major
// tensor (tensor_w x tensor_h, stride tensor_w) into contiguous LDS.
// D# packing per ISA §8.3 (group0) / §8.4 (group1); groups 2-3 zero (2D).
// This toolchain exposes the 6-arg builtin: (g0, g1, g2, g3, g4, cpol).
__device__ __forceinline__ void tdm_load_2d(unsigned lds_addr, const void* gaddr,
                                            unsigned tensor_w, unsigned tensor_h,
                                            unsigned tile_w, unsigned tile_h) {
  const unsigned long long ga = (unsigned long long)gaddr;
  uint4v g0;
  g0[0] = 1u;                                            // count=1, user mode
  g0[1] = lds_addr;                                      // lds_addr [63:32]
  g0[2] = (unsigned)(ga & 0xffffffffu);                  // global_addr lo
  g0[3] = (unsigned)((ga >> 32) & 0x01ffffffu) | (2u << 30);  // ga hi | type=2
  int8v g1;
  g1[0] = (int)(1u << 16);                               // data_size=1 (2B)
  g1[1] = (int)((tensor_w & 0xffffu) << 16);             // tensor_dim0[15:0]
  g1[2] = (int)((tensor_w >> 16) | ((tensor_h & 0xffffu) << 16));
  g1[3] = (int)((tensor_h >> 16) | (tile_w << 16));      // tile_dim0
  g1[4] = (int)tile_h;                                   // tile_dim1, dim2=0
  g1[5] = (int)tensor_w;                                 // dim0_stride lo
  g1[6] = 0;
  g1[7] = 0;
  const int4v gz4 = {};
  const int8v gz8 = {};
  __builtin_amdgcn_tensor_load_to_lds(g0, g1, gz4, gz4, gz8, 0);
}
#endif

// Fragment loader (bf16 row in global or LDS). ISA 16-bit striping:
// lane half owns K {8h..8h+7} and {16+8h..16+8h+7} of a K=32 slab.
__device__ __forceinline__ v16bf load_frag_bf16(const bf16* __restrict__ row,
                                                int kbase, int half) {
  const v8bf x = *(const v8bf*)(row + kbase + 8 * half);
  const v8bf y = *(const v8bf*)(row + kbase + 16 + 8 * half);
  v16bf r;
#pragma unroll
  for (int i = 0; i < 8; ++i) { r[i] = x[i]; r[8 + i] = y[i]; }
  return r;
}

// ---------------------------------------------------------------------------
// Kernel 0: fp32 -> bf16 conversion pass (8 elems/thread).
// ---------------------------------------------------------------------------
__global__ __launch_bounds__(256) void cvt_bf16_kernel(
    const float* __restrict__ src, bf16* __restrict__ dst, int n8) {
  const int i = blockIdx.x * 256 + threadIdx.x;
  if (i >= n8) return;
  const v4f a = ((const v4f*)src)[2 * i];
  const v4f b = ((const v4f*)src)[2 * i + 1];
  v8bf o;
#pragma unroll
  for (int j = 0; j < 4; ++j) { o[j] = (bf16)a[j]; o[4 + j] = (bf16)b[j]; }
  ((v8bf*)dst)[i] = o;
}

// ---------------------------------------------------------------------------
// Shared GEMM tile machinery: block 128 thr (4 waves), block tile 128x64,
// wave tile 32x64 (8 accumulators), K-step 64, TDM double-buffered staging
// (fallback: per-lane async copies). 12 LDS loads per 8 WMMAs per K=32 slab.
// ---------------------------------------------------------------------------
struct GemmCtx {
  int tid, wave, lane, half, ln, row0, col0;
};

// Stage A(128x64) / B(64x64) bf16 tiles for k-slab k0 into buffer `buf`.
template <bool kUseTdm>
__device__ __forceinline__ void stage_tiles(const GemmCtx& c,
                                            bf16 (*As)[128 * 64],
                                            bf16 (*Bs)[64 * 64], int buf,
                                            const bf16* Asrc, unsigned Ah,
                                            const bf16* Bsrc, unsigned Bh,
                                            int k0) {
#if HAS_TDM
  if (kUseTdm) {
    if (c.wave == 0) {
      tdm_load_2d(lds_off(&As[buf][0]), Asrc + (size_t)c.row0 * Dm + k0,
                  Dm, Ah, 64, 128);
      tdm_load_2d(lds_off(&Bs[buf][0]), Bsrc + (size_t)c.col0 * Dm + k0,
                  Dm, Bh, 64, 64);
    }
    return;
  }
#endif
#pragma unroll
  for (int i = 0; i < 8; ++i) {          // A: 128 rows x 128B = 1024 chunks
    const int ch = c.tid + i * 128, r = ch >> 3, cc = ch & 7;
    async_b128(lds_off(&As[buf][r * 64 + cc * 8]),
               Asrc + (size_t)(c.row0 + r) * Dm + k0 + cc * 8);
  }
#pragma unroll
  for (int i = 0; i < 4; ++i) {          // B: 64 rows x 128B = 512 chunks
    const int ch = c.tid + i * 128, r = ch >> 3, cc = ch & 7;
    async_b128(lds_off(&Bs[buf][r * 64 + cc * 8]),
               Bsrc + (size_t)(c.col0 + r) * Dm + k0 + cc * 8);
  }
}

template <bool kUseTdm>
__device__ __forceinline__ void stage_wait(const GemmCtx& c, int pending) {
#if HAS_TDM
  if (kUseTdm) {
    if (c.wave == 0) {
      if (pending) { WAIT_TENSOR(2); } else { WAIT_TENSOR(0); }
    }
    return;
  }
#endif
  if (pending) { WAIT_ASYNC(12); } else { WAIT_ASYNC(0); }
}

// K-loop producing 2x4 accumulators (wave tile 32x64).
template <bool kUseTdm>
__device__ __forceinline__ void gemm_mainloop(const GemmCtx& c,
                                              bf16 (*As)[128 * 64],
                                              bf16 (*Bs)[64 * 64],
                                              const bf16* Asrc, unsigned Ah,
                                              const bf16* Bsrc, unsigned Bh,
                                              v8f acc[2][4]) {
  stage_tiles<kUseTdm>(c, As, Bs, 0, Asrc, Ah, Bsrc, Bh, 0);
  for (int ks = 0; ks < Dm / 64; ++ks) {
    const int buf = ks & 1;
    const int more = (ks + 1 < Dm / 64);
    if (more) stage_tiles<kUseTdm>(c, As, Bs, buf ^ 1, Asrc, Ah, Bsrc, Bh,
                                   (ks + 1) * 64);
    stage_wait<kUseTdm>(c, more);
    __syncthreads();

#pragma unroll
    for (int kk = 0; kk < 2; ++kk) {
      const v16bf a0 = load_frag_bf16(&As[buf][(c.wave * 32 + c.ln) * 64],
                                      kk * 32, c.half);
      const v16bf a1 = load_frag_bf16(&As[buf][(c.wave * 32 + 16 + c.ln) * 64],
                                      kk * 32, c.half);
#pragma unroll
      for (int t = 0; t < 4; ++t) {
        const v16bf bfr = load_frag_bf16(&Bs[buf][(t * 16 + c.ln) * 64],
                                         kk * 32, c.half);
        acc[0][t] = wmma_bf16(a0, bfr, acc[0][t]);   // B reused by both WMMAs
        acc[1][t] = wmma_bf16(a1, bfr, acc[1][t]);
      }
    }
    __syncthreads();
  }
}

// ---------------------------------------------------------------------------
// Kernel 1: QKV projection.  Y = Xb @ Wb^T + bias.
// transpose_out: 0 -> out[(b*H+h)*S+s][d] ; 1 -> out[(b*H+h)*D+d][s] (for V)
// ---------------------------------------------------------------------------
__global__ __launch_bounds__(128) void qkv_proj_kernel(
    const bf16* __restrict__ Xb, const bf16* __restrict__ Wb,
    const float* __restrict__ bias, bf16* __restrict__ out,
    int transpose_out) {
  __shared__ bf16 As[2][128 * 64];
  __shared__ bf16 Bs[2][64 * 64];

  GemmCtx c;
  c.tid = threadIdx.x; c.wave = c.tid >> 5;
  c.lane = c.tid & 31; c.half = c.lane >> 4; c.ln = c.lane & 15;
  c.row0 = blockIdx.x * 128; c.col0 = blockIdx.y * 64;

  v8f acc[2][4];
  const v8f vzero = {};
#pragma unroll
  for (int rr = 0; rr < 2; ++rr)
#pragma unroll
    for (int t = 0; t < 4; ++t) acc[rr][t] = vzero;

  gemm_mainloop<true>(c, As, Bs, Xb, Bsz * Ssz, Wb, Dm, acc);

#pragma unroll
  for (int rr = 0; rr < 2; ++rr) {
#pragma unroll
    for (int t = 0; t < 4; ++t) {
      const int col = c.col0 + t * 16 + c.ln;
      const int h = col >> 6, d = col & 63;
      const float bv = bias[col];
#pragma unroll
      for (int r = 0; r < 8; ++r) {
        const int row = c.row0 + c.wave * 32 + rr * 16 + r + 8 * c.half;
        const int b_ = row >> 11, s_ = row & (Ssz - 1);
        const float v = acc[rr][t][r] + bv;
        if (transpose_out)
          out[((size_t)(b_ * Hn + h) * Dh + d) * Ssz + s_] = (bf16)v;
        else
          out[((size_t)(b_ * Hn + h) * Ssz + s_) * Dh + d] = (bf16)v;
      }
    }
  }
}

// ---------------------------------------------------------------------------
// Kernel 2: flash attention per (b, h, 64-row q-tile); K/V slabs staged in LDS
// with async copies and double buffering; 4 waves x 16 q-rows.
// ---------------------------------------------------------------------------
__global__ __launch_bounds__(128) void attn_kernel(
    const bf16* __restrict__ qb, const bf16* __restrict__ kb,
    const bf16* __restrict__ vT, const unsigned char* __restrict__ mask,
    bf16* __restrict__ ctx) {
  __shared__ bf16 Ks[2][32 * 64];    // 32 k-rows x 64 d
  __shared__ bf16 Vs[2][64 * 32];    // 64 d-rows x 32 s
  __shared__ bf16 plds[4][16 * 32];  // per-wave P staging

  const int tid  = threadIdx.x;
  const int wave = tid >> 5, lane = tid & 31;
  const int half = lane >> 4, ln = lane & 15;

  const int b  = blockIdx.z;
  const int h  = blockIdx.y;
  const int bh = b * Hn + h;
  const int q0 = blockIdx.x * 64 + wave * 16;

  auto stageKV = [&](int buf, int kc) {  // 4 async instrs / wave
#pragma unroll
    for (int i = 0; i < 2; ++i) {        // K slab: 32 x 128B
      const int c = tid + i * 128, r = c >> 3, cc = c & 7;
      async_b128(lds_off(&Ks[buf][r * 64 + cc * 8]),
                 kb + ((size_t)bh * Ssz + kc + r) * Dh + cc * 8);
    }
#pragma unroll
    for (int i = 0; i < 2; ++i) {        // V slab: 64 x 64B
      const int c = tid + i * 128, r = c >> 2, cc = c & 3;
      async_b128(lds_off(&Vs[buf][r * 32 + cc * 8]),
                 vT + ((size_t)bh * Dh + r) * Ssz + kc + cc * 8);
    }
  };

  const bf16* qrow = qb + ((size_t)bh * Ssz + q0 + ln) * Dh;
  const v16bf qa0 = load_frag_bf16(qrow, 0, half);
  const v16bf qa1 = load_frag_bf16(qrow, 32, half);

  float mrow[8], lrow[8];
  v8f oacc[4];
  const v8f vzero = {};
#pragma unroll
  for (int t = 0; t < 4; ++t) oacc[t] = vzero;
#pragma unroll
  for (int r = 0; r < 8; ++r) { mrow[r] = -1e30f; lrow[r] = 0.0f; }

  const unsigned char* mbase = mask + (size_t)b * Ssz * Ssz;
  bf16* pw = &plds[wave][0];

  stageKV(0, 0);
  for (int kc = 0; kc < Ssz; kc += 32) {
    const int buf = (kc >> 5) & 1;
    if (kc + 32 < Ssz) { stageKV(buf ^ 1, kc + 32); WAIT_ASYNC(4); }
    else               { WAIT_ASYNC(0); }
    __syncthreads();

    // ---- S = (Q K^T)/8 for 16x32 slab ----
    v8f s0 = vzero, s1 = vzero;
    {
      const bf16* k0r = &Ks[buf][ln * 64];
      s0 = wmma_bf16(qa0, load_frag_bf16(k0r, 0, half), s0);
      s0 = wmma_bf16(qa1, load_frag_bf16(k0r, 32, half), s0);
      const bf16* k1r = &Ks[buf][(16 + ln) * 64];
      s1 = wmma_bf16(qa0, load_frag_bf16(k1r, 0, half), s1);
      s1 = wmma_bf16(qa1, load_frag_bf16(k1r, 32, half), s1);
    }

    float pv0[8], pv1[8], tmax[8];
#pragma unroll
    for (int r = 0; r < 8; ++r) {
      const int row = q0 + r + 8 * half;
      float x0 = s0[r] * 0.125f;
      float x1 = s1[r] * 0.125f;
      if (mbase[(size_t)row * Ssz + kc + ln])      x0 = -1e9f;
      if (mbase[(size_t)row * Ssz + kc + 16 + ln]) x1 = -1e9f;
      pv0[r] = x0; pv1[r] = x1;
      tmax[r] = fmaxf(x0, x1);
    }
#pragma unroll
    for (int off = 1; off < 16; off <<= 1)
#pragma unroll
      for (int r = 0; r < 8; ++r)
        tmax[r] = fmaxf(tmax[r], __shfl_xor(tmax[r], off, 32));

    float alpha[8], psum[8];
#pragma unroll
    for (int r = 0; r < 8; ++r) {
      const float mnew = fmaxf(mrow[r], tmax[r]);
      alpha[r] = __expf(mrow[r] - mnew);
      mrow[r] = mnew;
      pv0[r] = __expf(pv0[r] - mnew);
      pv1[r] = __expf(pv1[r] - mnew);
      psum[r] = pv0[r] + pv1[r];
    }
#pragma unroll
    for (int off = 1; off < 16; off <<= 1)
#pragma unroll
      for (int r = 0; r < 8; ++r)
        psum[r] += __shfl_xor(psum[r], off, 32);
#pragma unroll
    for (int r = 0; r < 8; ++r) lrow[r] = lrow[r] * alpha[r] + psum[r];
#pragma unroll
    for (int t = 0; t < 4; ++t)
#pragma unroll
      for (int r = 0; r < 8; ++r) oacc[t][r] *= alpha[r];

    // ---- C-layout P -> row-major LDS -> A-fragment (in-order DS) ----
#pragma unroll
    for (int r = 0; r < 8; ++r) {
      pw[(r + 8 * half) * 32 + ln]      = (bf16)pv0[r];
      pw[(r + 8 * half) * 32 + 16 + ln] = (bf16)pv1[r];
    }
    const v16bf pa = load_frag_bf16(pw + ln * 32, 0, half);

    // ---- O += P V ----
#pragma unroll
    for (int t = 0; t < 4; ++t)
      oacc[t] = wmma_bf16(pa, load_frag_bf16(&Vs[buf][(t * 16 + ln) * 32], 0, half),
                          oacc[t]);
    __syncthreads();
  }

#pragma unroll
  for (int t = 0; t < 4; ++t) {
    const int col = h * Dh + t * 16 + ln;
#pragma unroll
    for (int r = 0; r < 8; ++r) {
      const int row = q0 + r + 8 * half;
      ctx[((size_t)b * Ssz + row) * Dm + col] = (bf16)(oacc[t][r] / lrow[r]);
    }
  }
}

// ---------------------------------------------------------------------------
// Kernel 3: output projection + bias + residual -> fp32 pre-LN (TDM-staged).
// ---------------------------------------------------------------------------
__global__ __launch_bounds__(128) void out_proj_kernel(
    const bf16* __restrict__ ctxb, const bf16* __restrict__ Wob,
    const float* __restrict__ bo, const float* __restrict__ resid,
    float* __restrict__ pre) {
  __shared__ bf16 As[2][128 * 64];
  __shared__ bf16 Bs[2][64 * 64];

  GemmCtx c;
  c.tid = threadIdx.x; c.wave = c.tid >> 5;
  c.lane = c.tid & 31; c.half = c.lane >> 4; c.ln = c.lane & 15;
  c.row0 = blockIdx.x * 128; c.col0 = blockIdx.y * 64;

  v8f acc[2][4];
  const v8f vzero = {};
#pragma unroll
  for (int rr = 0; rr < 2; ++rr)
#pragma unroll
    for (int t = 0; t < 4; ++t) acc[rr][t] = vzero;

  gemm_mainloop<true>(c, As, Bs, ctxb, Bsz * Ssz, Wob, Dm, acc);

#pragma unroll
  for (int rr = 0; rr < 2; ++rr) {
#pragma unroll
    for (int t = 0; t < 4; ++t) {
      const int col = c.col0 + t * 16 + c.ln;
      const float bv = bo[col];
#pragma unroll
      for (int r = 0; r < 8; ++r) {
        const int row = c.row0 + c.wave * 32 + rr * 16 + r + 8 * c.half;
        pre[(size_t)row * Dm + col] =
            acc[rr][t][r] + bv + resid[(size_t)row * Dm + col];
      }
    }
  }
}

// ---------------------------------------------------------------------------
// Kernel 4: per-row layer norm (1024 elems, 256 thr x 4).
// ---------------------------------------------------------------------------
__global__ __launch_bounds__(256) void ln_kernel(
    const float* __restrict__ x, const float* __restrict__ gamma,
    const float* __restrict__ beta, float* __restrict__ out) {
  __shared__ float sred[8], ssred[8];
  const int tid = threadIdx.x;
  const int lane = tid & 31;
  const int w = tid >> 5;
  const size_t row = blockIdx.x;

  const v4f v = ((const v4f*)(x + row * Dm))[tid];
  float s  = v[0] + v[1] + v[2] + v[3];
  float ss = v[0] * v[0] + v[1] * v[1] + v[2] * v[2] + v[3] * v[3];
#pragma unroll
  for (int off = 1; off < 32; off <<= 1) {
    s  += __shfl_xor(s, off, 32);
    ss += __shfl_xor(ss, off, 32);
  }
  if (lane == 0) { sred[w] = s; ssred[w] = ss; }
  __syncthreads();
  float st = 0.0f, sst = 0.0f;
#pragma unroll
  for (int i = 0; i < 8; ++i) { st += sred[i]; sst += ssred[i]; }

  const float mu   = st * (1.0f / Dm);
  const float var  = sst * (1.0f / Dm) - mu * mu;
  const float rinv = rsqrtf(var + 1e-5f);

  const v4f gv = ((const v4f*)gamma)[tid];
  const v4f bv = ((const v4f*)beta)[tid];
  v4f o;
#pragma unroll
  for (int j = 0; j < 4; ++j) o[j] = (v[j] - mu) * rinv * gv[j] + bv[j];
  ((v4f*)(out + row * Dm))[tid] = o;
}

// ---------------------------------------------------------------------------
// Launch
// ---------------------------------------------------------------------------
extern "C" void kernel_launch(void* const* d_in, const int* in_sizes, int n_in,
                              void* d_out, int out_size, void* d_ws,
                              size_t ws_size, hipStream_t stream) {
  (void)in_sizes; (void)n_in; (void)out_size; (void)ws_size;

  const float* Q = (const float*)d_in[0];
  const float* K = (const float*)d_in[1];
  const float* V = (const float*)d_in[2];
  const unsigned char* mask = (const unsigned char*)d_in[3];
  const float* Wq = (const float*)d_in[4];
  const float* bq = (const float*)d_in[5];
  const float* Wk = (const float*)d_in[6];
  const float* bk = (const float*)d_in[7];
  const float* Wv = (const float*)d_in[8];
  const float* bv = (const float*)d_in[9];
  const float* Wo = (const float*)d_in[10];
  const float* bo = (const float*)d_in[11];
  const float* lng = (const float*)d_in[12];
  const float* lnb = (const float*)d_in[13];
  float* out = (float*)d_out;

  // Workspace (bytes): activations/weights in bf16, pre-LN in f32. ~152 MB.
  const size_t MB = 1024 * 1024;
  char* w = (char*)d_ws;
  bf16* Xq  = (bf16*)(w + 0 * MB);     // 16 MB  [B*S,1024]
  bf16* Xk  = (bf16*)(w + 16 * MB);
  bf16* Xv  = (bf16*)(w + 32 * MB);
  bf16* Wqb = (bf16*)(w + 48 * MB);    // 2 MB each
  bf16* Wkb = (bf16*)(w + 50 * MB);
  bf16* Wvb = (bf16*)(w + 52 * MB);
  bf16* Wob = (bf16*)(w + 54 * MB);
  bf16* qbf = (bf16*)(w + 56 * MB);    // 16 MB [BH,S,64]
  bf16* kbf = (bf16*)(w + 72 * MB);
  bf16* vTb = (bf16*)(w + 88 * MB);    // [BH,64,S]
  bf16* ctx = (bf16*)(w + 104 * MB);   // 16 MB [B*S,1024]
  float* pre = (float*)(w + 120 * MB); // 32 MB

  const int nAct8 = (Bsz * Ssz * Dm) / 8;  // 1,048,576
  const int nW8   = (Dm * Dm) / 8;         // 131,072
  cvt_bf16_kernel<<<dim3(nAct8 / 256), dim3(256), 0, stream>>>(Q, Xq, nAct8);
  cvt_bf16_kernel<<<dim3(nAct8 / 256), dim3(256), 0, stream>>>(K, Xk, nAct8);
  cvt_bf16_kernel<<<dim3(nAct8 / 256), dim3(256), 0, stream>>>(V, Xv, nAct8);
  cvt_bf16_kernel<<<dim3(nW8 / 256), dim3(256), 0, stream>>>(Wq, Wqb, nW8);
  cvt_bf16_kernel<<<dim3(nW8 / 256), dim3(256), 0, stream>>>(Wk, Wkb, nW8);
  cvt_bf16_kernel<<<dim3(nW8 / 256), dim3(256), 0, stream>>>(Wv, Wvb, nW8);
  cvt_bf16_kernel<<<dim3(nW8 / 256), dim3(256), 0, stream>>>(Wo, Wob, nW8);

  const dim3 blk(128);
  const dim3 gP((Bsz * Ssz) / 128, Dm / 64);  // 64 x 16
  qkv_proj_kernel<<<gP, blk, 0, stream>>>(Xq, Wqb, bq, qbf, 0);
  qkv_proj_kernel<<<gP, blk, 0, stream>>>(Xk, Wkb, bk, kbf, 0);
  qkv_proj_kernel<<<gP, blk, 0, stream>>>(Xv, Wvb, bv, vTb, 1);

  const dim3 gA(Ssz / 64, Hn, Bsz);           // 32 x 16 x 4
  attn_kernel<<<gA, blk, 0, stream>>>(qbf, kbf, vTb, mask, ctx);

  out_proj_kernel<<<gP, blk, 0, stream>>>(ctx, Wob, bo, Q, pre);

  ln_kernel<<<dim3(Bsz * Ssz), dim3(256), 0, stream>>>(pre, lng, lnb, out);
}
